// MEDA_40381282517352
// MI455X (gfx1250) — compile-verified, matrix operand order checked by reading, not run on hardware
//
#include <hip/hip_runtime.h>

typedef __attribute__((ext_vector_type(16))) __bf16 v16bf;
typedef __attribute__((ext_vector_type(8)))  __bf16 v8bf;
typedef __attribute__((ext_vector_type(8)))  float  v8f;

#define DEV __device__ __forceinline__

// ---------------- shape constants (B=8, C=128, H=W=128, ps=16, step=14) -----
#define BB   8
#define CC   128
#define HH   128
#define WW   128
#define HWS  16384           // H*W
#define PS   16
#define STEP 14
#define NHT  9               // tiles per axis
#define NPB  81              // patches per batch
#define NPAT 648             // total patches
#define TT   256             // tokens per patch
#define HID  256

DEV v8f vzero8() { v8f z;
#pragma unroll
  for (int i = 0; i < 8; ++i) z[i] = 0.0f; return z; }

DEV v8f wmma_bf16(v16bf a, v16bf b, v8f c) {
  return __builtin_amdgcn_wmma_f32_16x16x32_bf16(false, a, false, b, (short)0, c,
                                                 false, false);
}

// Load one lane's 16 bf16 A/B-fragment values for a 16x32 slab row.
// Row layout per ISA: VGPR0..3 hold K = [8*half, 8*half+8), VGPR4..7 hold
// K = [16+8*half, 16+8*half+8); element order inside v16bf matches.
DEV v16bf frag_load(const __bf16* row, int half) {
  v8bf lo = *(const v8bf*)(row + 8 * half);
  v8bf hi = *(const v8bf*)(row + 16 + 8 * half);
  v16bf f;
#pragma unroll
  for (int i = 0; i < 8; ++i) { f[i] = lo[i]; f[i + 8] = hi[i]; }
  return f;
}

DEV float gelu_f(float x) { return 0.5f * x * (1.0f + erff(x * 0.70710678118654752f)); }
DEV float wred_sum(float v) {
#pragma unroll
  for (int d = 1; d < 32; d <<= 1) v += __shfl_xor(v, d, 32);
  return v;
}

// ---------------------------------------------------------------------------
// weight prep: fp32 (K,N) -> bf16 (N,K)   and plain fp32 -> bf16 copy
// ---------------------------------------------------------------------------
__global__ void conv_w_t(const float* __restrict__ src, __bf16* __restrict__ dst,
                         int K, int N) {
  int i = blockIdx.x * blockDim.x + threadIdx.x;
  if (i >= K * N) return;
  int n = i / K, k = i - n * K;
  dst[i] = (__bf16)src[k * N + n];
}
__global__ void conv_w(const float* __restrict__ src, __bf16* __restrict__ dst, int cnt) {
  int i = blockIdx.x * blockDim.x + threadIdx.x;
  if (i < cnt) dst[i] = (__bf16)src[i];
}

// ---------------------------------------------------------------------------
// 2x2 average pool:  low[b,c,i,j]
// ---------------------------------------------------------------------------
__global__ void lowpass_kernel(const float* __restrict__ x, float* __restrict__ low) {
  int idx = blockIdx.x * blockDim.x + threadIdx.x;           // BB*CC*64*64
  if (idx >= BB * CC * 64 * 64) return;
  int j = idx & 63, i = (idx >> 6) & 63, bc = idx >> 12;
  const float* xp = x + ((size_t)bc << 14) + ((i * 2) << 7) + (j * 2);
  low[idx] = 0.25f * (xp[0] + xp[1] + xp[WW] + xp[WW + 1]);
}

// ---------------------------------------------------------------------------
// hf = x - bilinear_up(low)    (half-pixel centers, edge clamp)
// ---------------------------------------------------------------------------
__global__ void hf_kernel(const float* __restrict__ x, const float* __restrict__ low,
                          float* __restrict__ hf) {
  int idx = blockIdx.x * blockDim.x + threadIdx.x;           // BB*CC*HH*WW
  if (idx >= BB * CC * HWS) return;
  int w = idx & 127, h = (idx >> 7) & 127, bc = idx >> 14;
  int kh = h >> 1, kw = w >> 1;
  int h2 = (h & 1) ? ((kh + 1 < 64) ? kh + 1 : 63) : ((kh > 0) ? kh - 1 : 0);
  int w2 = (w & 1) ? ((kw + 1 < 64) ? kw + 1 : 63) : ((kw > 0) ? kw - 1 : 0);
  const float* lp = low + ((size_t)bc << 12);
  float v = 0.5625f * lp[kh * 64 + kw] + 0.1875f * lp[kh * 64 + w2] +
            0.1875f * lp[h2 * 64 + kw] + 0.0625f * lp[h2 * 64 + w2];
  hf[idx] = x[idx] - v;
}

// ---------------------------------------------------------------------------
// sgate[b,h,w] = sigmoid( sum_c |hf| * gw[c] + gb )   one wave per pixel
// ---------------------------------------------------------------------------
__global__ void gate_kernel(const float* __restrict__ hf, const float* __restrict__ gw,
                            const float* __restrict__ gb, float* __restrict__ sg) {
  int pix = blockIdx.x * 8 + (threadIdx.x >> 5);             // BB*HWS pixels
  int lane = threadIdx.x & 31;
  if (pix >= BB * HWS) return;
  int b = pix >> 14, hw = pix & 16383;
  size_t base = ((size_t)b * CC << 14) + hw;
  float s = 0.0f;
#pragma unroll
  for (int i = 0; i < 4; ++i) {
    int c = lane + 32 * i;
    s += fabsf(hf[base + ((size_t)c << 14)]) * gw[c];
  }
  s = wred_sum(s);
  if (lane == 0) sg[pix] = 1.0f / (1.0f + __expf(-(s + gb[0])));
}

// ---------------------------------------------------------------------------
// gbuf[(b,hw),c] = bf16(gelu(dwconv3x3_dil2(hf)))   pixel-major for GEMM
// ---------------------------------------------------------------------------
__global__ void dw3_kernel(const float* __restrict__ hf, const float* __restrict__ w9,
                           __bf16* __restrict__ gbuf) {
  int idx = blockIdx.x * blockDim.x + threadIdx.x;           // NCHW order
  if (idx >= BB * CC * HWS) return;
  int w = idx & 127, h = (idx >> 7) & 127, c = (idx >> 14) & 127, b = idx >> 21;
  float s = 0.0f;
#pragma unroll
  for (int dy = 0; dy < 3; ++dy) {
    int ny = h + 2 * dy - 2;
    if (ny < 0 || ny >= HH) continue;
#pragma unroll
    for (int dx = 0; dx < 3; ++dx) {
      int nx = w + 2 * dx - 2;
      if (nx < 0 || nx >= WW) continue;
      s += hf[(((size_t)(b * CC + c)) << 14) + (ny << 7) + nx] * w9[c * 9 + dy * 3 + dx];
    }
  }
  gbuf[(((size_t)b << 14) + (h << 7) + w) * CC + c] = (__bf16)gelu_f(s);
}

// ---------------------------------------------------------------------------
// generic bf16 WMMA GEMM:  D = act(A[M,K] @ Wt[N,K]^T + bias) (+epilogue)
// block = 256 threads = 8 waves; each wave -> 16x64 block (4 accumulators,
// A fragment reused across 4 WMMAs per K-chunk); grid (M/128, N/64).
// B fragments are batch-loaded before the WMMA chain so the four WMMAs
// issue back-to-back on independent accumulators.
// ---------------------------------------------------------------------------
#define GM_PW   0
#define GM_QKV  1
#define GM_GELU 2
#define GM_PROJ 3
#define GM_FC2  4

__global__ void __launch_bounds__(256)
gemm_bf16_kernel(const __bf16* __restrict__ A, const __bf16* __restrict__ Wt,
                 int N, int K, int mode,
                 const float* __restrict__ bias,
                 const float* __restrict__ aux0, const float* __restrict__ aux1,
                 const float* __restrict__ aux2,
                 float* __restrict__ fout,
                 __bf16* __restrict__ bout, __bf16* __restrict__ bout2,
                 __bf16* __restrict__ bout3) {
  __shared__ __bf16 ldsW[64 * 256];                // up to 32 KB
  const int tid = threadIdx.x;
  const int wid = tid >> 5;
  const int lane = tid & 31;
  const int l16 = lane & 15;
  const int half = lane >> 4;
  const int nbase = blockIdx.y * 64;
  const int mbase = blockIdx.x * 128 + wid * 16;

  // stage 64 x K weight tile (rows are N; contiguous in memory)
  const __bf16* wsrc = Wt + (size_t)nbase * K;
  const int nunits = (64 * K) >> 3;
  for (int u = tid; u < nunits; u += 256)
    ((v8bf*)ldsW)[u] = ((const v8bf*)wsrc)[u];
  __syncthreads();

  v8f acc[4];
#pragma unroll
  for (int j = 0; j < 4; ++j) acc[j] = vzero8();
  const __bf16* arow = A + (size_t)(mbase + l16) * K;
  const __bf16* brow = ldsW + l16 * K;
  for (int kc = 0; kc < K; kc += 32) {
    __builtin_prefetch(arow + kc + 128, 0, 3);     // near-scope prefetch
    v16bf af = frag_load(arow + kc, half);
    v16bf bf[4];
#pragma unroll
    for (int j = 0; j < 4; ++j) bf[j] = frag_load(brow + j * 16 * K + kc, half);
#pragma unroll
    for (int j = 0; j < 4; ++j) acc[j] = wmma_bf16(af, bf[j], acc[j]);
  }

#pragma unroll
  for (int j = 0; j < 4; ++j) {
    const int n = nbase + j * 16 + l16;
    const float bv = bias ? bias[n] : 0.0f;
#pragma unroll
    for (int r = 0; r < 8; ++r) {
      const int m = mbase + r + 8 * half;
      float v = acc[j][r] + bv;
      switch (mode) {
        case GM_PW: {  // x1 = x + lth_scale * refined * sgate
          int b = m >> 14, hw = m & 16383;
          size_t idx = ((size_t)(b * CC + n) << 14) + hw;
          fout[idx] = aux0[idx] + aux1[n] * v * aux2[m];
        } break;
        case GM_QKV: {
          if (n < CC) bout[(size_t)m * CC + n] = (__bf16)v;
          else if (n < 2 * CC) bout2[(size_t)m * CC + (n - CC)] = (__bf16)v;
          else {  // v stored channel-major (patch, ch, tok)
            int p = m >> 8, t = m & 255;
            bout3[(((size_t)p * CC) + (n - 2 * CC)) * TT + t] = (__bf16)v;
          }
        } break;
        case GM_GELU: {
          bout[(size_t)m * N + n] = (__bf16)gelu_f(v);
        } break;
        case GM_PROJ: {  // + crop residual, overlap-add into summed image
          int p = m >> 8, t = m & 255;
          int b = p / NPB, pp = p % NPB, ih = pp / NHT, iw = pp % NHT;
          int h = ih * STEP + (t >> 4), w = iw * STEP + (t & 15);
          size_t idx = ((size_t)(b * CC + n) << 14) + (h << 7) + w;
          atomicAdd(&fout[idx], v + aux0[idx]);
        } break;
        case GM_FC2: {  // + residual, write final NCHW output
          int b = m >> 14, hw = m & 16383;
          size_t idx = ((size_t)(b * CC + n) << 14) + hw;
          fout[idx] = v + aux0[idx];
        } break;
      }
    }
  }
}

// ---------------------------------------------------------------------------
// LayerNorm over C=128 (one wave per token).
// mode 0: gather crop token from x1 (NCHW);  mode 1: token from x2 (NCHW)
// ---------------------------------------------------------------------------
__global__ void __launch_bounds__(256)
ln_kernel(const float* __restrict__ xin, const float* __restrict__ g,
          const float* __restrict__ bta, __bf16* __restrict__ yout,
          int ntok, int mode) {
  int tok = blockIdx.x * 8 + (threadIdx.x >> 5);
  int lane = threadIdx.x & 31;
  if (tok >= ntok) return;
  size_t base;
  if (mode == 0) {
    int p = tok >> 8, t = tok & 255;
    int b = p / NPB, pp = p % NPB, ih = pp / NHT, iw = pp % NHT;
    int h = ih * STEP + (t >> 4), w = iw * STEP + (t & 15);
    base = ((size_t)(b * CC) << 14) + (h << 7) + w;
  } else {
    int b = tok >> 14, hw = tok & 16383;
    base = ((size_t)(b * CC) << 14) + hw;
  }
  float v[4];
  float s = 0.0f;
#pragma unroll
  for (int i = 0; i < 4; ++i) {
    int c = lane + 32 * i;
    v[i] = xin[base + ((size_t)c << 14)];
    s += v[i];
  }
  s = wred_sum(s);
  float mu = s * (1.0f / 128.0f);
  float vs = 0.0f;
#pragma unroll
  for (int i = 0; i < 4; ++i) { float d = v[i] - mu; vs += d * d; }
  vs = wred_sum(vs);
  float rs = rsqrtf(vs * (1.0f / 128.0f) + 1e-6f);
#pragma unroll
  for (int i = 0; i < 4; ++i) {
    int c = lane + 32 * i;
    yout[(size_t)tok * CC + c] = (__bf16)((v[i] - mu) * rs * g[c] + bta[c]);
  }
}

// ---------------------------------------------------------------------------
// PE: pe = dwconv3x3(q as patch image) + pe_b     (token-major bf16)
// ---------------------------------------------------------------------------
__global__ void pe_kernel(const __bf16* __restrict__ qb, const float* __restrict__ pew,
                          const float* __restrict__ peb_, __bf16* __restrict__ pe) {
  int idx = blockIdx.x * blockDim.x + threadIdx.x;  // NPAT*TT*CC
  if (idx >= NPAT * TT * CC) return;
  int c = idx & 127, t = (idx >> 7) & 255, p = idx >> 15;
  int py = t >> 4, px = t & 15;
  float s = 0.0f;
#pragma unroll
  for (int dy = 0; dy < 3; ++dy) {
    int ny = py + dy - 1;
    if (ny < 0 || ny >= PS) continue;
#pragma unroll
    for (int dx = 0; dx < 3; ++dx) {
      int nx = px + dx - 1;
      if (nx < 0 || nx >= PS) continue;
      s += (float)qb[(((size_t)p << 8) + (ny << 4) + nx) * CC + c] *
           pew[c * 9 + dy * 3 + dx];
    }
  }
  pe[idx] = (__bf16)(s + peb_[c]);
}

// ---------------------------------------------------------------------------
// Flash-style windowed attention. 1 block per patch, 16 waves, each wave
// owns one 16-token query tile. Q/K token-major bf16, V channel-major bf16.
// Epilogue: (O/l + pe) * z  -> proj input (token-major bf16)
// ---------------------------------------------------------------------------
__global__ void __launch_bounds__(512)
attn_kernel(const __bf16* __restrict__ qb, const __bf16* __restrict__ kb,
            const __bf16* __restrict__ vb, const __bf16* __restrict__ pe,
            const __bf16* __restrict__ zb, __bf16* __restrict__ pin) {
  __shared__ __bf16 ldsP[16][512];                 // per-wave 16x32 P staging
  const int p = blockIdx.x;
  const int wid = threadIdx.x >> 5;
  const int lane = threadIdx.x & 31;
  const int l16 = lane & 15;
  const int half = lane >> 4;
  __bf16* lp = &ldsP[wid][0];

  const __bf16* qrow = qb + ((size_t)(p * TT + wid * 16 + l16)) * CC;
  v16bf aQ[4];
#pragma unroll
  for (int c4 = 0; c4 < 4; ++c4) aQ[c4] = frag_load(qrow + 32 * c4, half);

  v8f O[8];
  float mi[8], li[8];
#pragma unroll
  for (int ct = 0; ct < 8; ++ct) O[ct] = vzero8();
#pragma unroll
  for (int r = 0; r < 8; ++r) { mi[r] = -1e30f; li[r] = 0.0f; }
  const float scl = 0.08838834764831845f;          // 1/sqrt(128)

  for (int kp = 0; kp < 8; ++kp) {                 // 32 keys per iter
    const int k0 = kp * 32;
    v8f s0 = vzero8(), s1 = vzero8();
    const __bf16* krow0 = kb + ((size_t)(p * TT + k0 + l16)) * CC;
    const __bf16* krow1 = krow0 + 16 * CC;
#pragma unroll
    for (int c4 = 0; c4 < 4; ++c4) {
      s0 = wmma_bf16(aQ[c4], frag_load(krow0 + 32 * c4, half), s0);
      s1 = wmma_bf16(aQ[c4], frag_load(krow1 + 32 * c4, half), s1);
    }
    float p0[8], p1[8];
#pragma unroll
    for (int r = 0; r < 8; ++r) {
      float a = s0[r] * scl, b2 = s1[r] * scl;
      float mx = fmaxf(a, b2);
      mx = fmaxf(mx, __shfl_xor(mx, 1, 32));
      mx = fmaxf(mx, __shfl_xor(mx, 2, 32));
      mx = fmaxf(mx, __shfl_xor(mx, 4, 32));
      mx = fmaxf(mx, __shfl_xor(mx, 8, 32));
      float mnew = fmaxf(mi[r], mx);
      float corr = __expf(mi[r] - mnew);
      float e0 = __expf(a - mnew), e1 = __expf(b2 - mnew);
      float rs = e0 + e1;
      rs += __shfl_xor(rs, 1, 32);
      rs += __shfl_xor(rs, 2, 32);
      rs += __shfl_xor(rs, 4, 32);
      rs += __shfl_xor(rs, 8, 32);
      li[r] = li[r] * corr + rs;
      mi[r] = mnew;
      p0[r] = e0; p1[r] = e1;
#pragma unroll
      for (int ct = 0; ct < 8; ++ct) O[ct][r] *= corr;
    }
    // C-layout -> A-layout restripe of P through LDS
#pragma unroll
    for (int r = 0; r < 8; ++r) {
      int row = r + 8 * half;
      lp[row * 32 + l16] = (__bf16)p0[r];
      lp[row * 32 + 16 + l16] = (__bf16)p1[r];
    }
    asm volatile("s_wait_dscnt 0" ::: "memory");
    v16bf aP = frag_load(lp + l16 * 32, half);
#pragma unroll
    for (int ct = 0; ct < 8; ++ct) {
      const __bf16* vrow = vb + (((size_t)p * CC) + ct * 16 + l16) * TT + k0;
      O[ct] = wmma_bf16(aP, frag_load(vrow, half), O[ct]);
    }
  }
#pragma unroll
  for (int ct = 0; ct < 8; ++ct) {
    int ch = ct * 16 + l16;
#pragma unroll
    for (int r = 0; r < 8; ++r) {
      int tok = p * TT + wid * 16 + r + 8 * half;
      size_t ix = (size_t)tok * CC + ch;
      float val = O[ct][r] / li[r];
      val = (val + (float)pe[ix]) * (float)zb[ix];
      pin[ix] = (__bf16)val;
    }
  }
}

// ---------------------------------------------------------------------------
// overlap-add divisor (analytic): f(i)=2 on overlap bands [14k,14k+2)
// ---------------------------------------------------------------------------
__global__ void divide_kernel(float* __restrict__ summed) {
  int idx = blockIdx.x * blockDim.x + threadIdx.x;
  if (idx >= BB * CC * HWS) return;
  int w = idx & 127, h = (idx >> 7) & 127;
  float fh = (h >= 14 && h < 114 && (h % 14) < 2) ? 0.5f : 1.0f;
  float fw = (w >= 14 && w < 114 && (w % 14) < 2) ? 0.5f : 1.0f;
  summed[idx] *= fh * fw;
}

// ---------------------------------------------------------------------------
// MLP dwconv5x5 (pad 2) on token-major hidden:  y2 = y + conv(y) + b
// ---------------------------------------------------------------------------
__global__ void dw5_kernel(const __bf16* __restrict__ y, const float* __restrict__ w25,
                           const float* __restrict__ db, __bf16* __restrict__ y2) {
  int idx = blockIdx.x * blockDim.x + threadIdx.x;  // BB*HWS*HID
  if (idx >= BB * HWS * HID) return;
  int cc = idx & 255, hw = (idx >> 8) & 16383, b = idx >> 22;
  int h = hw >> 7, w = hw & 127;
  float s = 0.0f;
#pragma unroll
  for (int dy = 0; dy < 5; ++dy) {
    int nh = h + dy - 2;
    if (nh < 0 || nh >= HH) continue;
#pragma unroll
    for (int dx = 0; dx < 5; ++dx) {
      int nw = w + dx - 2;
      if (nw < 0 || nw >= WW) continue;
      s += (float)y[(((size_t)b << 14) + (nh << 7) + nw) * HID + cc] *
           w25[cc * 25 + dy * 5 + dx];
    }
  }
  y2[idx] = (__bf16)((float)y[idx] + s + db[cc]);
}

// ---------------------------------------------------------------------------
extern "C" void kernel_launch(void* const* d_in, const int* in_sizes, int n_in,
                              void* d_out, int out_size, void* d_ws, size_t ws_size,
                              hipStream_t stream) {
  const float* x        = (const float*)d_in[0];
  const float* lth_dw_w = (const float*)d_in[2];
  const float* lth_pw_w = (const float*)d_in[3];
  const float* lth_pw_b = (const float*)d_in[4];
  const float* lth_gw   = (const float*)d_in[5];
  const float* lth_gb   = (const float*)d_in[6];
  const float* lth_sc   = (const float*)d_in[7];
  const float* ln1_g    = (const float*)d_in[8];
  const float* ln1_b    = (const float*)d_in[9];
  const float* ln2_g    = (const float*)d_in[10];
  const float* ln2_b    = (const float*)d_in[11];
  const float* qkv_w    = (const float*)d_in[12];
  const float* gate_w   = (const float*)d_in[13];
  const float* gate_b   = (const float*)d_in[14];
  const float* proj_w   = (const float*)d_in[15];
  const float* pe_w     = (const float*)d_in[16];
  const float* pe_b     = (const float*)d_in[17];
  const float* fc1_w    = (const float*)d_in[18];
  const float* fc1_b    = (const float*)d_in[19];
  const float* dw_w     = (const float*)d_in[20];
  const float* dw_b     = (const float*)d_in[21];
  const float* fc2_w    = (const float*)d_in[22];
  const float* fc2_b    = (const float*)d_in[23];
  float* out = (float*)d_out;

  char* ws = (char*)d_ws;
  size_t off = 0;
  auto alloc = [&](size_t bytes) -> void* {
    void* p = ws + off;
    off += (bytes + 255) & ~(size_t)255;
    return p;
  };
  const size_t NCHW = (size_t)BB * CC * HWS;          // 16.7M elems
  const size_t NTOKP = (size_t)NPAT * TT;             // 165888
  const size_t NTOKI = (size_t)BB * HWS;              // 131072

  float*  f_low = (float*)alloc((size_t)BB * CC * 64 * 64 * 4);
  float*  f_hf  = (float*)alloc(NCHW * 4);
  float*  f_sg  = (float*)alloc(NTOKI * 4);
  __bf16* b_g   = (__bf16*)alloc(NCHW * 2);
  float*  f_x1  = (float*)alloc(NCHW * 4);
  __bf16* b_y1  = (__bf16*)alloc(NTOKP * CC * 2);
  __bf16* b_q   = (__bf16*)alloc(NTOKP * CC * 2);
  __bf16* b_k   = (__bf16*)alloc(NTOKP * CC * 2);
  __bf16* b_v   = (__bf16*)alloc(NTOKP * CC * 2);
  __bf16* b_z   = (__bf16*)alloc(NTOKP * CC * 2);
  __bf16* b_pe  = (__bf16*)alloc(NTOKP * CC * 2);
  __bf16* b_pin = (__bf16*)alloc(NTOKP * CC * 2);
  float*  f_sum = (float*)alloc(NCHW * 4);            // becomes x2 after divide
  __bf16* b_xln = (__bf16*)alloc(NTOKI * CC * 2);
  __bf16* b_fc1 = (__bf16*)alloc(NTOKI * HID * 2);
  __bf16* b_y2  = (__bf16*)alloc(NTOKI * HID * 2);
  __bf16* b_wq  = (__bf16*)alloc(384 * 128 * 2);
  __bf16* b_wg  = (__bf16*)alloc(128 * 128 * 2);
  __bf16* b_wp  = (__bf16*)alloc(128 * 128 * 2);
  __bf16* b_w1  = (__bf16*)alloc(256 * 128 * 2);
  __bf16* b_w2  = (__bf16*)alloc(128 * 256 * 2);
  __bf16* b_pw  = (__bf16*)alloc(128 * 128 * 2);

  // ---- weight prep (bf16, N-major x K) ----
  conv_w_t<<<dim3(192), 256, 0, stream>>>(qkv_w,  b_wq, 128, 384);
  conv_w_t<<<dim3(64),  256, 0, stream>>>(gate_w, b_wg, 128, 128);
  conv_w_t<<<dim3(64),  256, 0, stream>>>(proj_w, b_wp, 128, 128);
  conv_w_t<<<dim3(128), 256, 0, stream>>>(fc1_w,  b_w1, 128, 256);
  conv_w_t<<<dim3(128), 256, 0, stream>>>(fc2_w,  b_w2, 256, 128);
  conv_w  <<<dim3(64),  256, 0, stream>>>(lth_pw_w, b_pw, 128 * 128);

  // ---- LTH branch ----
  lowpass_kernel<<<dim3(16384), 256, 0, stream>>>(x, f_low);
  hf_kernel<<<dim3(65536), 256, 0, stream>>>(x, f_low, f_hf);
  gate_kernel<<<dim3(16384), 256, 0, stream>>>(f_hf, lth_gw, lth_gb, f_sg);
  dw3_kernel<<<dim3(65536), 256, 0, stream>>>(f_hf, lth_dw_w, b_g);
  gemm_bf16_kernel<<<dim3(1024, 2), 256, 0, stream>>>(
      b_g, b_pw, 128, 128, GM_PW, lth_pw_b, x, lth_sc, f_sg, f_x1,
      nullptr, nullptr, nullptr);

  // ---- patch attention ----
  ln_kernel<<<dim3(20736), 256, 0, stream>>>(f_x1, ln1_g, ln1_b, b_y1,
                                             (int)NTOKP, 0);
  gemm_bf16_kernel<<<dim3(1296, 6), 256, 0, stream>>>(
      b_y1, b_wq, 384, 128, GM_QKV, nullptr, nullptr, nullptr, nullptr,
      nullptr, b_q, b_k, b_v);
  gemm_bf16_kernel<<<dim3(1296, 2), 256, 0, stream>>>(
      b_y1, b_wg, 128, 128, GM_GELU, gate_b, nullptr, nullptr, nullptr,
      nullptr, b_z, nullptr, nullptr);
  pe_kernel<<<dim3(82944), 256, 0, stream>>>(b_q, pe_w, pe_b, b_pe);
  hipMemsetAsync(f_sum, 0, NCHW * 4, stream);
  attn_kernel<<<dim3(NPAT), 512, 0, stream>>>(b_q, b_k, b_v, b_pe, b_z, b_pin);
  gemm_bf16_kernel<<<dim3(1296, 2), 256, 0, stream>>>(
      b_pin, b_wp, 128, 128, GM_PROJ, nullptr, f_x1, nullptr, nullptr, f_sum,
      nullptr, nullptr, nullptr);
  divide_kernel<<<dim3(65536), 256, 0, stream>>>(f_sum);   // f_sum == x2 now

  // ---- conv-MLP ----
  ln_kernel<<<dim3(16384), 256, 0, stream>>>(f_sum, ln2_g, ln2_b, b_xln,
                                             (int)NTOKI, 1);
  gemm_bf16_kernel<<<dim3(1024, 4), 256, 0, stream>>>(
      b_xln, b_w1, 256, 128, GM_GELU, fc1_b, nullptr, nullptr, nullptr,
      nullptr, b_fc1, nullptr, nullptr);
  dw5_kernel<<<dim3(131072), 256, 0, stream>>>(b_fc1, dw_w, dw_b, b_y2);
  gemm_bf16_kernel<<<dim3(1024, 2), 256, 0, stream>>>(
      b_y2, b_w2, 128, 256, GM_FC2, fc2_b, f_sum, nullptr, nullptr, out,
      nullptr, nullptr, nullptr);

  (void)in_sizes; (void)n_in; (void)out_size; (void)ws_size;
}
